// GNNEncoder_9904194584978
// MI455X (gfx1250) — compile-verified, matrix operand order checked by reading, not run on hardware
//
#include <hip/hip_runtime.h>
#include <hip/hip_bf16.h>

#define HD 256      // hidden dim
#define NVV 192     // nodes
#define NBB 2       // batch
#define NLL 12      // layers
#define NEDGE (NBB*NVV*NVV)   // 73728 edge rows
#define NNODE (NBB*NVV)       // 384 node rows
#define EPSV 1e-5f

typedef __attribute__((ext_vector_type(16))) __bf16 v16bf;
typedef __attribute__((ext_vector_type(8)))  float  v8f;

union FragU { v16bf v; uint4 q[2]; };

__device__ __forceinline__ unsigned short f2bf(float f) {
  unsigned int u = __float_as_uint(f);
  u += 0x7FFFu + ((u >> 16) & 1u);          // round-to-nearest-even
  return (unsigned short)(u >> 16);
}
__device__ __forceinline__ float bf2f(unsigned short h) {
  return __uint_as_float(((unsigned int)h) << 16);
}

// Load one 16x32 bf16 A/B fragment from a row-major [rows x 256] bf16 matrix.
// A-layout (ISA 7.12.2): lane<16 -> M=lane, K={0..7,16..23}; lane>=16 -> M=lane-16, K={8..15,24..31}.
// B is identical with N playing the role of M, since B[k,n] = W[n,k] for y = x @ W.T.
__device__ __forceinline__ v16bf load_frag(const unsigned short* base, int row0, int k0, int lane) {
  int half = lane >> 4, lr = lane & 15;
  const unsigned short* p = base + (size_t)(row0 + lr) * HD + k0 + half * 8;
  FragU f;
  f.q[0] = *(const uint4*)(p);
  f.q[1] = *(const uint4*)(p + 16);
  return f.v;
}

__device__ __forceinline__ v8f wmma_bf16(v16bf a, v16bf b, v8f c) {
  return __builtin_amdgcn_wmma_f32_16x16x32_bf16(false, a, false, b, (short)0, c, false, false);
}

// ---------------- weight cast ----------------
__global__ void cast_bf16_kernel(const float* __restrict__ src, unsigned short* __restrict__ dst, int n) {
  int i = blockIdx.x * 256 + threadIdx.x;
  if (i < n) dst[i] = f2bf(src[i]);
}

// ---------------- node position embedding: h = sincos(x) @ node_w.T + node_b ----------------
__global__ void node_embed_kernel(const float* __restrict__ x, const float* __restrict__ node_w,
                                  const float* __restrict__ node_b,
                                  float* __restrict__ hF, unsigned short* __restrict__ hB) {
  __shared__ float feat[HD];
  int t = threadIdx.x;
  int row = blockIdx.x;                      // b*V + v
  float coord = x[row * 2 + (t < 128 ? 0 : 1)];
  int i = t & 127;
  float inv = __powf(10000.f, -(2.f * (float)(i >> 1)) / 128.f);
  float p = coord * 6.28318530717958647692f * inv;
  feat[t] = (i & 1) ? __cosf(p) : __sinf(p);
  __syncthreads();
  float acc = node_b[t];
  const float* wrow = node_w + (size_t)t * HD;
  for (int k = 0; k < HD; ++k) acc += feat[k] * wrow[k];
  size_t idx = (size_t)row * HD + t;
  hF[idx] = acc;
  hB[idx] = f2bf(acc);
}

// ---------------- edge embedding: e = sincos(graph/dte) @ edge_w.T + edge_b (WMMA) ----------------
__global__ void edge_embed_kernel(const float* __restrict__ graph, const unsigned short* __restrict__ wE,
                                  const float* __restrict__ eb,
                                  float* __restrict__ eF, unsigned short* __restrict__ eB) {
  __shared__ __attribute__((aligned(32))) unsigned short featS[32 * HD];  // 16 KB bf16 features
  int tid = threadIdx.x;
  int lane = tid & 31, wv = tid >> 5;
  int mw = wv & 1, nw = wv >> 1;
  int m0 = blockIdx.x * 32;

  float inv = __powf(10000.f, -(2.f * (float)(tid >> 1)) / 256.f);
  bool odd = (tid & 1);
  for (int r = 0; r < 32; ++r) {
    float g = graph[m0 + r];
    float p = g * inv;
    featS[r * HD + tid] = f2bf(odd ? __cosf(p) : __sinf(p));
  }
  __syncthreads();

  v8f acc[4];
  for (int t = 0; t < 4; ++t) for (int r = 0; r < 8; ++r) acc[t][r] = 0.f;
#pragma unroll
  for (int k0 = 0; k0 < HD; k0 += 32) {
    v16bf a = load_frag(featS, mw * 16, k0, lane);
#pragma unroll
    for (int t = 0; t < 4; ++t) {
      v16bf b = load_frag(wE, nw * 64 + t * 16, k0, lane);
      acc[t] = wmma_bf16(a, b, acc[t]);
    }
  }
  int half = lane >> 4, lr = lane & 15;
  for (int t = 0; t < 4; ++t)
    for (int r = 0; r < 8; ++r) {
      int row = m0 + mw * 16 + r + 8 * half;
      int n = nw * 64 + t * 16 + lr;
      float v = acc[t][r] + eb[n];
      size_t idx = (size_t)row * HD + n;
      eF[idx] = v;
      eB[idx] = f2bf(v);
    }
}

// ---------------- per-layer: 4 node GEMMs Uh/Vh/Ah/Bh = h @ W.T + b (WMMA) ----------------
__global__ void node_gemms_kernel(const unsigned short* __restrict__ hB,
                                  const unsigned short* w0, const unsigned short* w1,
                                  const unsigned short* w2, const unsigned short* w3,
                                  const float* b0, const float* b1, const float* b2, const float* b3,
                                  float* o0, float* o1, float* o2, float* o3) {
  int sel = blockIdx.y;
  const unsigned short* W = sel == 0 ? w0 : sel == 1 ? w1 : sel == 2 ? w2 : w3;
  const float* bias       = sel == 0 ? b0 : sel == 1 ? b1 : sel == 2 ? b2 : b3;
  float* out              = sel == 0 ? o0 : sel == 1 ? o1 : sel == 2 ? o2 : o3;

  int tid = threadIdx.x, lane = tid & 31, wv = tid >> 5;
  int mw = wv & 1, nw = wv >> 1;
  int m0 = blockIdx.x * 32;
  v8f acc[4];
  for (int t = 0; t < 4; ++t) for (int r = 0; r < 8; ++r) acc[t][r] = 0.f;
#pragma unroll
  for (int k0 = 0; k0 < HD; k0 += 32) {
    v16bf a = load_frag(hB, m0 + mw * 16, k0, lane);
#pragma unroll
    for (int t = 0; t < 4; ++t) {
      v16bf b = load_frag(W, nw * 64 + t * 16, k0, lane);
      acc[t] = wmma_bf16(a, b, acc[t]);
    }
  }
  int half = lane >> 4, lr = lane & 15;
  for (int t = 0; t < 4; ++t)
    for (int r = 0; r < 8; ++r) {
      int row = m0 + mw * 16 + r + 8 * half;
      int n = nw * 64 + t * 16 + lr;
      out[(size_t)row * HD + n] = acc[t][r] + bias[n];
    }
}

// ---------------- per-layer edge GEMM + fused epilogue ----------------
// e_new = e@Cw.T + cb + Ah[b,j] + Bh[b,i];  s = bf16(e_new);
// u = bf16( silu( ln2( relu( ln1(e_new) ) ) ) )
__global__ void layer_edge_kernel(const unsigned short* __restrict__ eB,
                                  const unsigned short* __restrict__ wC, const float* __restrict__ cb,
                                  const float* __restrict__ Ah, const float* __restrict__ Bh,
                                  const float* __restrict__ neg, const float* __restrict__ neb,
                                  const float* __restrict__ plg, const float* __restrict__ plb,
                                  unsigned short* __restrict__ sB, unsigned short* __restrict__ uB) {
  __shared__ float eS[32 * 257];           // padded rows to dodge bank conflicts
  __shared__ float st[4 * 32];             // m1, r1, m2, r2 per local row
  __shared__ float negS[HD], nebS[HD];
  int tid = threadIdx.x, lane = tid & 31, wv = tid >> 5;
  int mw = wv & 1, nw = wv >> 1;
  int m0 = blockIdx.x * 32;
  negS[tid] = neg[tid];
  nebS[tid] = neb[tid];

  v8f acc[4];
  for (int t = 0; t < 4; ++t) for (int r = 0; r < 8; ++r) acc[t][r] = 0.f;
#pragma unroll
  for (int k0 = 0; k0 < HD; k0 += 32) {
    v16bf a = load_frag(eB, m0 + mw * 16, k0, lane);
#pragma unroll
    for (int t = 0; t < 4; ++t) {
      v16bf b = load_frag(wC, nw * 64 + t * 16, k0, lane);
      acc[t] = wmma_bf16(a, b, acc[t]);
    }
  }
  int half = lane >> 4, lr = lane & 15;
  for (int t = 0; t < 4; ++t)
    for (int r = 0; r < 8; ++r) {
      int lrow = mw * 16 + r + 8 * half;
      int m = m0 + lrow;
      int n = nw * 64 + t * 16 + lr;
      int b = m / (NVV * NVV);
      int rem = m - b * (NVV * NVV);
      int i = rem / NVV, j = rem - i * NVV;
      float v = acc[t][r] + cb[n] + Ah[(size_t)(b * NVV + j) * HD + n] + Bh[(size_t)(b * NVV + i) * HD + n];
      eS[lrow * 257 + n] = v;
    }
  __syncthreads();

  if (tid < 32) {
    const float* row = &eS[tid * 257];
    float s = 0.f;
    for (int n = 0; n < HD; ++n) s += row[n];
    float m1 = s * (1.f / HD);
    float vv = 0.f;
    for (int n = 0; n < HD; ++n) { float d = row[n] - m1; vv += d * d; }
    float r1 = rsqrtf(vv * (1.f / HD) + EPSV);
    float s2 = 0.f, ss2 = 0.f;
    for (int n = 0; n < HD; ++n) {
      float y = fmaxf((row[n] - m1) * r1 * negS[n] + nebS[n], 0.f);
      s2 += y; ss2 += y * y;
    }
    float m2 = s2 * (1.f / HD);
    float var2 = ss2 * (1.f / HD) - m2 * m2;
    st[tid] = m1; st[32 + tid] = r1; st[64 + tid] = m2; st[96 + tid] = rsqrtf(var2 + EPSV);
  }
  __syncthreads();

  int n = tid;
  float gne = negS[n], bne = nebS[n];
  float gpl = plg[n], bpl = plb[n];
  for (int r = 0; r < 32; ++r) {
    float v = eS[r * 257 + n];
    size_t idx = (size_t)(m0 + r) * HD + n;
    sB[idx] = f2bf(v);
    float y = fmaxf((v - st[r]) * st[32 + r] * gne + bne, 0.f);
    float z = (y - st[64 + r]) * st[96 + r] * gpl + bpl;
    float u = z / (1.f + __expf(-z));           // silu
    uB[idx] = f2bf(u);
  }
}

// ---------------- per-layer node update: h += relu(ln(Uh + Σ_j sigmoid(e_new)·Vh)) ----------------
__global__ void node_update_kernel(const unsigned short* __restrict__ sB, const float* __restrict__ Vh,
                                   const float* __restrict__ Uh,
                                   const float* __restrict__ nhg, const float* __restrict__ nhb,
                                   float* __restrict__ hF, unsigned short* __restrict__ hB) {
  __shared__ float red[HD];
  int t = threadIdx.x;
  int bi = blockIdx.x;                 // b*V + i
  int b = bi / NVV, i = bi - b * NVV;
  size_t sbase = ((size_t)b * NVV * NVV + (size_t)i * NVV) * HD + t;
  size_t vbase = ((size_t)b * NVV) * HD + t;
  float acc = Uh[(size_t)bi * HD + t];
  for (int j = 0; j < NVV; ++j) {
    float xg = bf2f(sB[sbase + (size_t)j * HD]);
    float gate = 1.f / (1.f + __expf(-xg));
    acc += gate * Vh[vbase + (size_t)j * HD];
  }
  red[t] = acc; __syncthreads();
  for (int s = 128; s > 0; s >>= 1) { if (t < s) red[t] += red[t + s]; __syncthreads(); }
  float mean = red[0] * (1.f / HD); __syncthreads();
  float d = acc - mean;
  red[t] = d * d; __syncthreads();
  for (int s = 128; s > 0; s >>= 1) { if (t < s) red[t] += red[t + s]; __syncthreads(); }
  float var = red[0] * (1.f / HD);
  float hn = fmaxf(d * rsqrtf(var + EPSV) * nhg[t] + nhb[t], 0.f);
  size_t idx = (size_t)bi * HD + t;
  float hv = hF[idx] + hn;
  hF[idx] = hv;
  hB[idx] = f2bf(hv);
}

// ---------------- per-layer pl GEMM: e += u @ pl_w.T + pl_lb (WMMA, fused residual) ----------------
__global__ void layer_pl_kernel(const unsigned short* __restrict__ uB, const unsigned short* __restrict__ wPL,
                                const float* __restrict__ plbb,
                                float* __restrict__ eF, unsigned short* __restrict__ eB) {
  int tid = threadIdx.x, lane = tid & 31, wv = tid >> 5;
  int mw = wv & 1, nw = wv >> 1;
  int m0 = blockIdx.x * 32;
  v8f acc[4];
  for (int t = 0; t < 4; ++t) for (int r = 0; r < 8; ++r) acc[t][r] = 0.f;
#pragma unroll
  for (int k0 = 0; k0 < HD; k0 += 32) {
    v16bf a = load_frag(uB, m0 + mw * 16, k0, lane);
#pragma unroll
    for (int t = 0; t < 4; ++t) {
      v16bf b = load_frag(wPL, nw * 64 + t * 16, k0, lane);
      acc[t] = wmma_bf16(a, b, acc[t]);
    }
  }
  int half = lane >> 4, lr = lane & 15;
  for (int t = 0; t < 4; ++t)
    for (int r = 0; r < 8; ++r) {
      int row = m0 + mw * 16 + r + 8 * half;
      int n = nw * 64 + t * 16 + lr;
      size_t idx = (size_t)row * HD + n;
      float nv = eF[idx] + acc[t][r] + plbb[n];
      eF[idx] = nv;
      eB[idx] = f2bf(nv);
    }
}

// ---------------- GroupNorm stats: deterministic two-stage reduction ----------------
__global__ void gn_partial_kernel(const float* __restrict__ eF, float* __restrict__ part) {
  __shared__ float r1[256], r2[256];
  int t = threadIdx.x;
  int bg = blockIdx.x >> 4, ck = blockIdx.x & 15;
  int b = bg >> 5, g = bg & 31;
  float s = 0.f, ss = 0.f;
  for (int idx = ck * 256 + t; idx < 294912; idx += 4096) {      // (H/G)*V*V elements per (b,g)
    int p = idx >> 3, hh = idx & 7;
    float v = eF[((size_t)b * (NVV * NVV) + p) * HD + g * 8 + hh];
    s += v; ss += v * v;
  }
  r1[t] = s; r2[t] = ss; __syncthreads();
  for (int st = 128; st > 0; st >>= 1) {
    if (t < st) { r1[t] += r1[t + st]; r2[t] += r2[t + st]; }
    __syncthreads();
  }
  if (t == 0) { part[blockIdx.x * 2] = r1[0]; part[blockIdx.x * 2 + 1] = r2[0]; }
}

__global__ void gn_finish_kernel(const float* __restrict__ part, float* __restrict__ gstat) {
  int bg = threadIdx.x;
  if (bg < 64) {
    float s = 0.f, ss = 0.f;
    for (int c = 0; c < 16; ++c) { s += part[(bg * 16 + c) * 2]; ss += part[(bg * 16 + c) * 2 + 1]; }
    float mean = s * (1.f / 294912.f);
    float var = ss * (1.f / 294912.f) - mean * mean;
    gstat[bg * 2] = mean;
    gstat[bg * 2 + 1] = rsqrtf(var + EPSV);
  }
}

// ---------------- out head: relu(groupnorm(e)) . conv_w + conv_b ----------------
__global__ void out_head_kernel(const float* __restrict__ eF, const float* __restrict__ gstat,
                                const float* __restrict__ gng, const float* __restrict__ gnb,
                                const float* __restrict__ cw, const float* __restrict__ cbias,
                                float* __restrict__ out) {
  __shared__ float r0[256], r1s[256];
  int t = threadIdx.x;
  int m = blockIdx.x;                        // b*V*V + i*V + j
  int b = m / (NVV * NVV);
  int rem = m - b * (NVV * NVV);
  float v = eF[(size_t)m * HD + t];
  int g = t >> 3;
  float mean = gstat[(b * 32 + g) * 2], rstd = gstat[(b * 32 + g) * 2 + 1];
  float y = fmaxf((v - mean) * rstd * gng[t] + gnb[t], 0.f);
  r0[t] = y * cw[t];
  r1s[t] = y * cw[HD + t];
  __syncthreads();
  for (int s = 128; s > 0; s >>= 1) {
    if (t < s) { r0[t] += r0[t + s]; r1s[t] += r1s[t + s]; }
    __syncthreads();
  }
  if (t == 0) {
    out[(size_t)(b * 2 + 0) * (NVV * NVV) + rem] = r0[0] + cbias[0];
    out[(size_t)(b * 2 + 1) * (NVV * NVV) + rem] = r1s[0] + cbias[1];
  }
}

// ================= host launch =================
extern "C" void kernel_launch(void* const* d_in, const int* in_sizes, int n_in,
                              void* d_out, int out_size, void* d_ws, size_t ws_size,
                              hipStream_t stream) {
  const float* x      = (const float*)d_in[0];
  const float* graph  = (const float*)d_in[1];
  const float* node_w = (const float*)d_in[2];
  const float* node_b = (const float*)d_in[3];
  const float* edge_w = (const float*)d_in[4];
  const float* edge_b = (const float*)d_in[5];
  const float* Uw = (const float*)d_in[6];
  const float* Ub = (const float*)d_in[7];
  const float* Vw = (const float*)d_in[8];
  const float* Vb = (const float*)d_in[9];
  const float* Aw = (const float*)d_in[10];
  const float* Ab = (const float*)d_in[11];
  const float* Bw = (const float*)d_in[12];
  const float* Bb = (const float*)d_in[13];
  const float* Cw = (const float*)d_in[14];
  const float* Cb = (const float*)d_in[15];
  const float* nh_g = (const float*)d_in[16];
  const float* nh_b = (const float*)d_in[17];
  const float* ne_g = (const float*)d_in[18];
  const float* ne_b = (const float*)d_in[19];
  const float* pl_g = (const float*)d_in[20];
  const float* pl_b = (const float*)d_in[21];
  const float* pl_w = (const float*)d_in[22];
  const float* pl_lb = (const float*)d_in[23];
  const float* gn_g = (const float*)d_in[24];
  const float* gn_b = (const float*)d_in[25];
  const float* conv_w = (const float*)d_in[26];
  const float* conv_b = (const float*)d_in[27];
  (void)in_sizes; (void)n_in; (void)out_size; (void)ws_size;

  char* ws = (char*)d_ws;
  size_t off = 0;
  auto take = [&](size_t bytes) { char* p = ws + off; off = (off + bytes + 255) & ~(size_t)255; return p; };

  const size_t HH = (size_t)HD * HD;
  unsigned short* wU  = (unsigned short*)take(NLL * HH * 2);
  unsigned short* wV  = (unsigned short*)take(NLL * HH * 2);
  unsigned short* wA  = (unsigned short*)take(NLL * HH * 2);
  unsigned short* wB  = (unsigned short*)take(NLL * HH * 2);
  unsigned short* wC  = (unsigned short*)take(NLL * HH * 2);
  unsigned short* wPL = (unsigned short*)take(NLL * HH * 2);
  unsigned short* wE  = (unsigned short*)take(HH * 2);
  float* hF  = (float*)take((size_t)NNODE * HD * 4);
  unsigned short* hB = (unsigned short*)take((size_t)NNODE * HD * 2);
  float* UhP = (float*)take((size_t)NNODE * HD * 4);
  float* VhP = (float*)take((size_t)NNODE * HD * 4);
  float* AhP = (float*)take((size_t)NNODE * HD * 4);
  float* BhP = (float*)take((size_t)NNODE * HD * 4);
  float* eF  = (float*)take((size_t)NEDGE * HD * 4);
  unsigned short* eB = (unsigned short*)take((size_t)NEDGE * HD * 2);
  unsigned short* sB = (unsigned short*)take((size_t)NEDGE * HD * 2);
  unsigned short* uB = (unsigned short*)take((size_t)NEDGE * HD * 2);
  float* part  = (float*)take(1024 * 2 * 4);
  float* gstat = (float*)take(64 * 2 * 4);

  const int NW = NLL * (int)HH;
  int gcast = (NW + 255) / 256;
  cast_bf16_kernel<<<gcast, 256, 0, stream>>>(Uw, wU, NW);
  cast_bf16_kernel<<<gcast, 256, 0, stream>>>(Vw, wV, NW);
  cast_bf16_kernel<<<gcast, 256, 0, stream>>>(Aw, wA, NW);
  cast_bf16_kernel<<<gcast, 256, 0, stream>>>(Bw, wB, NW);
  cast_bf16_kernel<<<gcast, 256, 0, stream>>>(Cw, wC, NW);
  cast_bf16_kernel<<<gcast, 256, 0, stream>>>(pl_w, wPL, NW);
  cast_bf16_kernel<<<((int)HH + 255) / 256, 256, 0, stream>>>(edge_w, wE, (int)HH);

  node_embed_kernel<<<NNODE, 256, 0, stream>>>(x, node_w, node_b, hF, hB);
  edge_embed_kernel<<<NEDGE / 32, 256, 0, stream>>>(graph, wE, edge_b, eF, eB);

  for (int l = 0; l < NLL; ++l) {
    node_gemms_kernel<<<dim3(NNODE / 32, 4), 256, 0, stream>>>(
        hB, wU + l * HH, wV + l * HH, wA + l * HH, wB + l * HH,
        Ub + l * HD, Vb + l * HD, Ab + l * HD, Bb + l * HD,
        UhP, VhP, AhP, BhP);
    layer_edge_kernel<<<NEDGE / 32, 256, 0, stream>>>(
        eB, wC + l * HH, Cb + l * HD, AhP, BhP,
        ne_g + l * HD, ne_b + l * HD, pl_g + l * HD, pl_b + l * HD, sB, uB);
    node_update_kernel<<<NNODE, 256, 0, stream>>>(
        sB, VhP, UhP, nh_g + l * HD, nh_b + l * HD, hF, hB);
    layer_pl_kernel<<<NEDGE / 32, 256, 0, stream>>>(
        uB, wPL + l * HH, pl_lb + l * HD, eF, eB);
  }

  gn_partial_kernel<<<1024, 256, 0, stream>>>(eF, part);
  gn_finish_kernel<<<1, 64, 0, stream>>>(part, gstat);
  out_head_kernel<<<NEDGE, 256, 0, stream>>>(eF, gstat, gn_g, gn_b, conv_w, conv_b, (float*)d_out);
}
